// GRU_15934328668493
// MI455X (gfx1250) — matrix operand order repaired and checked by reading, not compiled, not run on hardware
//
#include <hip/hip_runtime.h>

// ---------------------------------------------------------------------------
// 2-layer GRU (torch semantics) for MI455X / gfx1250, bf16 WMMA + f32 accum.
//   B=64, T=256, I=1024, H=512
// Plan:
//   1) convert x + all 4 weight matrices to bf16 scratch
//   2) gx1 = x_bf @ w_ih1^T + b_ih1          (WMMA GEMM, M=16384 N=1536 K=1024)
//   3) persistent scan layer1 (grid barrier per t), emits relu(h1) as bf16
//   4) gx2 = h1_bf @ w_ih2^T + b_ih2          (WMMA GEMM, M=16384 N=3072 K=512)
//   5) persistent scan layer2, emits f32 output
// ---------------------------------------------------------------------------

typedef __bf16 bf16;
typedef __attribute__((ext_vector_type(8)))  __bf16 v8bf;
typedef __attribute__((ext_vector_type(16))) __bf16 v16bf;
typedef __attribute__((ext_vector_type(8)))  float  v8f;

#define LANE (threadIdx.x & 31)

// ---- WMMA fragment helpers (gfx1250 wave32 layouts, CDNA5 ISA) ----

// A fragment: 16x32 bf16 tile of row-major A[M,K], rows row0..row0+15, k0..k0+31.
// lanes 0-15:  halves 0-7 = K k0..k0+7,   halves 8-15 = K k0+16..k0+23
// lanes 16-31: halves 0-7 = K k0+8..k0+15, halves 8-15 = K k0+24..k0+31
__device__ __forceinline__ v16bf load_frag_a(const bf16* __restrict__ A, int ldA,
                                             int row0, int k0, int lane) {
    int r = row0 + (lane & 15);
    int koff = (lane & 16) ? 8 : 0;
    const bf16* p = A + (size_t)r * ldA + k0 + koff;
    v8bf lo = *(const v8bf*)(p);
    v8bf hi = *(const v8bf*)(p + 16);
    return __builtin_shufflevector(lo, hi, 0,1,2,3,4,5,6,7,8,9,10,11,12,13,14,15);
}

// B fragment: 32x16 bf16 tile, sourced from W[N,K] row-major (i.e. B = W^T).
// lane holds column n0+(lane&15); same K layout as the A fragment:
// lanes 0-15: K k0..k0+7, k0+16..k0+23; lanes 16-31: K k0+8..k0+15, k0+24..k0+31.
__device__ __forceinline__ v16bf load_frag_b(const bf16* __restrict__ W, int ldW,
                                             int col0, int k0, int lane) {
    int c = col0 + (lane & 15);
    int koff = (lane & 16) ? 8 : 0;
    const bf16* p = W + (size_t)c * ldW + k0 + koff;
    v8bf lo = *(const v8bf*)(p);
    v8bf hi = *(const v8bf*)(p + 16);
    return __builtin_shufflevector(lo, hi, 0,1,2,3,4,5,6,7,8,9,10,11,12,13,14,15);
}

__device__ __forceinline__ v8f wmma_bf16(v16bf a, v16bf b, v8f c) {
    return __builtin_amdgcn_wmma_f32_16x16x32_bf16(
        /*neg_a=*/false, a, /*neg_b=*/false, b,
        /*c_mod=*/(short)0, c, /*reuse_a=*/false, /*reuse_b=*/false);
}

__device__ __forceinline__ float sigmoid_f(float x) {
    return 1.0f / (1.0f + __expf(-x));
}
__device__ __forceinline__ float tanh_f(float x) {
    // tanh(x) = 1 - 2/(exp(2x)+1); saturates correctly at +-1.
    return 1.0f - 2.0f / (__expf(2.0f * x) + 1.0f);
}

// ---------------------------------------------------------------------------
// elementwise f32 -> bf16
__global__ void f32_to_bf16_kernel(const float* __restrict__ in,
                                   bf16* __restrict__ out, size_t n) {
    size_t i = (size_t)blockIdx.x * blockDim.x + threadIdx.x;
    size_t stride = (size_t)gridDim.x * blockDim.x;
    for (; i < n; i += stride) out[i] = (bf16)in[i];
}

__global__ void zero_kernel(float* __restrict__ p, size_t n) {
    size_t i = (size_t)blockIdx.x * blockDim.x + threadIdx.x;
    size_t stride = (size_t)gridDim.x * blockDim.x;
    for (; i < n; i += stride) p[i] = 0.0f;
}

// ---------------------------------------------------------------------------
// C[M,N] = A[M,K] @ W[N,K]^T + bias[N]    (A,W bf16; C f32)
// block = 256 threads (8 waves). Block tile 128x128.
// Wave (wr=wave>>1, wc=wave&1): 32 rows x 64 cols -> 2 A-frags, 4 B-frags, 8 WMMA.
__global__ __launch_bounds__(256) void gemm_bias_kernel(
    const bf16* __restrict__ A, const bf16* __restrict__ W,
    const float* __restrict__ bias, float* __restrict__ C,
    int M, int N, int K) {
    const int lane = LANE;
    const int wave = threadIdx.x >> 5;
    const int row0 = blockIdx.y * 128 + (wave >> 1) * 32;
    const int col0 = blockIdx.x * 128 + (wave & 1) * 64;

    v8f acc[2][4] = {};
    for (int k0 = 0; k0 < K; k0 += 32) {
        v16bf a0 = load_frag_a(A, K, row0,      k0, lane);
        v16bf a1 = load_frag_a(A, K, row0 + 16, k0, lane);
        v16bf b0 = load_frag_b(W, K, col0 +  0, k0, lane);
        v16bf b1 = load_frag_b(W, K, col0 + 16, k0, lane);
        v16bf b2 = load_frag_b(W, K, col0 + 32, k0, lane);
        v16bf b3 = load_frag_b(W, K, col0 + 48, k0, lane);
        acc[0][0] = wmma_bf16(a0, b0, acc[0][0]);
        acc[0][1] = wmma_bf16(a0, b1, acc[0][1]);
        acc[0][2] = wmma_bf16(a0, b2, acc[0][2]);
        acc[0][3] = wmma_bf16(a0, b3, acc[0][3]);
        acc[1][0] = wmma_bf16(a1, b0, acc[1][0]);
        acc[1][1] = wmma_bf16(a1, b1, acc[1][1]);
        acc[1][2] = wmma_bf16(a1, b2, acc[1][2]);
        acc[1][3] = wmma_bf16(a1, b3, acc[1][3]);
    }
    // C/D layout: VGPR i, lanes 0-15 -> row i, lanes 16-31 -> row i+8; col = lane&15
    const int cbase = col0 + (lane & 15);
    const int rhalf = (lane & 16) ? 8 : 0;
#pragma unroll
    for (int r = 0; r < 2; ++r) {
#pragma unroll
        for (int c = 0; c < 4; ++c) {
            int col = cbase + 16 * c;
            float bv = bias[col];
#pragma unroll
            for (int i = 0; i < 8; ++i) {
                int row = row0 + 16 * r + rhalf + i;
                C[(size_t)row * N + col] = acc[r][c][i] + bv;
            }
        }
    }
}

// ---------------------------------------------------------------------------
// Persistent GRU scan. One wave owns a (r,z,n) gate triple for a 16x16 h-block
// => nonlinearity + h-update are wave-local. Grid barrier per timestep.
// HD = hidden size (512 layer1, 1024 layer2). T=256, B=64 fixed.
template <int HD, bool RELU_BF16_OUT>
__global__ __launch_bounds__(256) void gru_scan_kernel(
    const float* __restrict__ gx,        // [B*T, 3*HD] = x-projection (+b_ih)
    const bf16*  __restrict__ Whh,       // [3*HD, HD] bf16
    const float* __restrict__ bhh,       // [3*HD]
    float*       __restrict__ hps,       // ping-pong f32 h, 2 * 64*HD
    bf16*        __restrict__ hbf,       // ping-pong bf16 h, 2 * 64*HD
    unsigned*    __restrict__ bar,       // grid barrier counter (zeroed)
    bf16*        __restrict__ out_bf,    // layer1: relu(h) bf16 [B*T, HD]
    float*       __restrict__ out_f)     // layer2: h f32 [B*T, HD]
{
    constexpr int T  = 256;
    constexpr int JT = HD / 16;          // gate-column tiles
    const int lane = LANE;
    const int w  = blockIdx.x * 8 + (threadIdx.x >> 5);
    const int m  = w / JT;               // batch row-tile 0..3
    const int jt = w % JT;
    const int rowA0 = m * 16;
    const int colR = jt * 16;
    const int colZ = HD + jt * 16;
    const int colN = 2 * HD + jt * 16;
    const unsigned nblocks = gridDim.x;

    const int j     = jt * 16 + (lane & 15);
    const int rhalf = (lane & 16) ? 8 : 0;
    const float bvr = bhh[j], bvz = bhh[HD + j], bvn = bhh[2 * HD + j];

    int cur = 0;
    for (int t = 0; t < T; ++t) {
        const bf16*  hb  = hbf + (size_t)cur       * (64 * HD);
        const float* hf  = hps + (size_t)cur       * (64 * HD);
        bf16*        hbn = hbf + (size_t)(cur ^ 1) * (64 * HD);
        float*       hfn = hps + (size_t)(cur ^ 1) * (64 * HD);

        v8f aR = {}, aZ = {}, aN = {};
#pragma unroll 4
        for (int k0 = 0; k0 < HD; k0 += 32) {
            v16bf a  = load_frag_a(hb,  HD, rowA0, k0, lane);
            v16bf bR = load_frag_b(Whh, HD, colR,  k0, lane);
            v16bf bZ = load_frag_b(Whh, HD, colZ,  k0, lane);
            v16bf bN = load_frag_b(Whh, HD, colN,  k0, lane);
            aR = wmma_bf16(a, bR, aR);
            aZ = wmma_bf16(a, bZ, aZ);
            aN = wmma_bf16(a, bN, aN);
        }
#pragma unroll
        for (int i = 0; i < 8; ++i) {
            int b = rowA0 + rhalf + i;
            size_t gxrow = ((size_t)b * T + t) * (size_t)(3 * HD);
            float xr = gx[gxrow + j];
            float xz = gx[gxrow + HD + j];
            float xn = gx[gxrow + 2 * HD + j];
            float r  = sigmoid_f(xr + aR[i] + bvr);
            float z  = sigmoid_f(xz + aZ[i] + bvz);
            float nn = tanh_f(xn + r * (aN[i] + bvn));
            float hp = hf[(size_t)b * HD + j];
            float hn = (1.0f - z) * nn + z * hp;
            hfn[(size_t)b * HD + j] = hn;
            hbn[(size_t)b * HD + j] = (bf16)hn;
            size_t orow = ((size_t)b * T + t) * (size_t)HD + j;
            if (RELU_BF16_OUT) out_bf[orow] = (bf16)fmaxf(hn, 0.0f);
            else               out_f[orow]  = hn;
        }
        // ---- device-scope grid barrier (monotonic counter) ----
        __threadfence();
        __syncthreads();
        if (threadIdx.x == 0) {
            atomicAdd(bar, 1u);
            unsigned target = nblocks * (unsigned)(t + 1);
            while (__hip_atomic_load(bar, __ATOMIC_RELAXED,
                                     __HIP_MEMORY_SCOPE_AGENT) < target)
                __builtin_amdgcn_s_sleep(2);
        }
        __syncthreads();
        __threadfence();
        cur ^= 1;
    }
}

// ---------------------------------------------------------------------------
extern "C" void kernel_launch(void* const* d_in, const int* in_sizes, int n_in,
                              void* d_out, int out_size, void* d_ws, size_t ws_size,
                              hipStream_t stream) {
    constexpr int B = 64, T = 256, I = 1024, H = 512;
    constexpr int M  = B * T;        // 16384
    constexpr int N1 = 3 * H;        // 1536
    constexpr int N2 = 3 * I;        // 3072

    const float* x     = (const float*)d_in[0];
    const float* w_ih1 = (const float*)d_in[1];
    const float* w_hh1 = (const float*)d_in[2];
    const float* b_ih1 = (const float*)d_in[3];
    const float* b_hh1 = (const float*)d_in[4];
    const float* w_ih2 = (const float*)d_in[5];
    const float* w_hh2 = (const float*)d_in[6];
    const float* b_ih2 = (const float*)d_in[7];
    const float* b_hh2 = (const float*)d_in[8];
    float* out = (float*)d_out;

    // ---- workspace layout (256B aligned) ----
    char* ws = (char*)d_ws;
    size_t off = 0;
    auto alloc = [&](size_t bytes) {
        void* p = ws + off;
        off = (off + bytes + 255) & ~(size_t)255;
        return p;
    };
    bf16*  w_ih1_bf = (bf16*)alloc((size_t)N1 * I * 2);
    bf16*  w_hh1_bf = (bf16*)alloc((size_t)N1 * H * 2);
    bf16*  w_ih2_bf = (bf16*)alloc((size_t)N2 * H * 2);
    bf16*  w_hh2_bf = (bf16*)alloc((size_t)N2 * I * 2);
    bf16*  x_bf     = (bf16*)alloc((size_t)M * I * 2);
    bf16*  h1_bf    = (bf16*)alloc((size_t)M * H * 2);
    float* gx1      = (float*)alloc((size_t)M * N1 * 4);
    float* gx2      = (float*)alloc((size_t)M * N2 * 4);
    // scan state: [hps 2*64*I f32 | hbf 2*64*I bf16 | counter]
    float*    hps = (float*)alloc((size_t)2 * 64 * I * 4);
    bf16*     hbf = (bf16*)alloc((size_t)2 * 64 * I * 2);
    unsigned* bar = (unsigned*)alloc(256);
    const size_t state_words =
        ((char*)bar + 256 - (char*)hps) / sizeof(float);

    auto cvt = [&](const float* src, bf16* dst, size_t n) {
        int blocks = (int)((n + 255) / 256);
        if (blocks > 4096) blocks = 4096;
        f32_to_bf16_kernel<<<blocks, 256, 0, stream>>>(src, dst, n);
    };

    // 1) conversions
    cvt(w_ih1, w_ih1_bf, (size_t)N1 * I);
    cvt(w_hh1, w_hh1_bf, (size_t)N1 * H);
    cvt(w_ih2, w_ih2_bf, (size_t)N2 * H);
    cvt(w_hh2, w_hh2_bf, (size_t)N2 * I);
    cvt(x,     x_bf,     (size_t)M * I);

    // 2) gx1 = x_bf @ w_ih1^T + b_ih1
    gemm_bias_kernel<<<dim3(N1 / 128, M / 128), 256, 0, stream>>>(
        x_bf, w_ih1_bf, b_ih1, gx1, M, N1, I);

    // 3) layer-1 scan: 4 row-tiles * 32 gate-col-tiles = 128 waves = 16 blocks
    zero_kernel<<<512, 256, 0, stream>>>(hps, state_words);
    gru_scan_kernel<H, true><<<16, 256, 0, stream>>>(
        gx1, w_hh1_bf, b_hh1, hps, hbf, bar, h1_bf, nullptr);

    // 4) gx2 = relu(h1)_bf @ w_ih2^T + b_ih2
    gemm_bias_kernel<<<dim3(N2 / 128, M / 128), 256, 0, stream>>>(
        h1_bf, w_ih2_bf, b_ih2, gx2, M, N2, H);

    // 5) layer-2 scan: 4 * 64 = 256 waves = 32 blocks
    zero_kernel<<<512, 256, 0, stream>>>(hps, state_words);
    gru_scan_kernel<I, false><<<32, 256, 0, stream>>>(
        gx2, w_hh2_bf, b_hh2, hps, hbf, bar, nullptr, out);

    (void)in_sizes; (void)n_in; (void)out_size; (void)ws_size;
}
